// MHA_73332271612200
// MI455X (gfx1250) — matrix operand order repaired and checked by reading, not compiled, not run on hardware
//
#include <hip/hip_runtime.h>
#include <hip/hip_bf16.h>

// MHA forward: B=4, N=1024, D=768, H=12, HD=64 (fp32 in/out).
// bf16 WMMA everywhere (f32 accumulate); bf16 intermediates; V tiles staged
// into LDS with GLOBAL_LOAD_ASYNC_TO_LDS_B128 (double-buffered, ASYNCcnt).

typedef __attribute__((ext_vector_type(16))) __bf16 v16bf;
typedef __attribute__((ext_vector_type(8)))  __bf16 v8bf;
typedef __attribute__((ext_vector_type(8)))  float  v8f;

#define BATCH 4
#define NSEQ  1024
#define DMODEL 768
#define NHEAD 12
#define HDIM  64
#define QKVD  (3 * DMODEL)   // 2304

#define WMMA_BF16(a, b, c)                                              \
  __builtin_amdgcn_wmma_f32_16x16x32_bf16(false, (a), false, (b),       \
                                          (short)0, (c), false, false)

// ---- one-shot f32 -> bf16 conversion (8 elems/thread, b128 in / b128 out)
__global__ void __launch_bounds__(256)
cvt_f32_to_bf16(const float* __restrict__ in, __bf16* __restrict__ out, int n) {
  const int i = (blockIdx.x * blockDim.x + threadIdx.x) * 8;
  if (i + 8 > n) return;
  const float4 a = *(const float4*)(in + i);
  const float4 b = *(const float4*)(in + i + 4);
  v8bf o;
  o[0] = (__bf16)a.x; o[1] = (__bf16)a.y; o[2] = (__bf16)a.z; o[3] = (__bf16)a.w;
  o[4] = (__bf16)b.x; o[5] = (__bf16)b.y; o[6] = (__bf16)b.z; o[7] = (__bf16)b.w;
  *(v8bf*)(out + i) = o;
}

// ---- bf16 fragment loaders (contiguous k = 16*half + s, same for A and B) -----------------
// A (16x32): lane l = row M=l&15; lanes 0-15 K=0..15, 16-31 K=16..31
__device__ __forceinline__ v16bf frag_a_bf(const __bf16* __restrict__ rowp,
                                           int k0, int lane) {
  const int kh = (lane >> 4) * 16;
  const v8bf lo = *(const v8bf*)(rowp + k0 + kh);
  const v8bf hi = *(const v8bf*)(rowp + k0 + kh + 8);
  v16bf f;
#pragma unroll
  for (int j = 0; j < 8; ++j) { f[j] = lo[j]; f[8 + j] = hi[j]; }
  return f;
}
// B (32x16): lane l = column N=l&15; lanes 0-15 K=0..15, 16-31 K=16..31
__device__ __forceinline__ v16bf frag_b_bf(const __bf16* __restrict__ colp,
                                           int k0, int lane) {
  const int kb = (lane >> 4) * 16;
  const v8bf lo = *(const v8bf*)(colp + k0 + kb);
  const v8bf hi = *(const v8bf*)(colp + k0 + kb + 8);
  v16bf f;
#pragma unroll
  for (int j = 0; j < 8; ++j) { f[j] = lo[j]; f[8 + j] = hi[j]; }
  return f;
}

// ---- GEMM: C[M,N] = A[M,K] * B[N,K]^T (+bias), bf16 in, CT out ---------
template <typename CT>
__global__ void __launch_bounds__(128)
wmma_gemm_nt(const __bf16* __restrict__ A, const __bf16* __restrict__ B,
             CT* __restrict__ C, const float* __restrict__ bias,
             int M, int K, int N) {
  const int lane = threadIdx.x & 31;
  const int wave = blockIdx.x * (blockDim.x >> 5) + (threadIdx.x >> 5);
  const int ntiles = N >> 5;
  const int tm = (wave / ntiles) << 5;
  const int tn = (wave % ntiles) << 5;
  if (tm >= M) return;

  const __bf16* arow0 = A + (size_t)(tm + (lane & 15)) * K;
  const __bf16* arow1 = arow0 + (size_t)16 * K;
  const __bf16* bcol0 = B + (size_t)(tn + (lane & 15)) * K;
  const __bf16* bcol1 = bcol0 + (size_t)16 * K;

  v8f acc00 = {}, acc01 = {}, acc10 = {}, acc11 = {};
  for (int k = 0; k < K; k += 32) {
    const v16bf a0 = frag_a_bf(arow0, k, lane);
    const v16bf a1 = frag_a_bf(arow1, k, lane);
    const v16bf b0 = frag_b_bf(bcol0, k, lane);
    const v16bf b1 = frag_b_bf(bcol1, k, lane);
    acc00 = WMMA_BF16(a0, b0, acc00);
    acc01 = WMMA_BF16(a0, b1, acc01);
    acc10 = WMMA_BF16(a1, b0, acc10);
    acc11 = WMMA_BF16(a1, b1, acc11);
  }

  const int n0  = tn + (lane & 15);
  const int hlf = lane >> 4;
  const float bi0 = bias ? bias[n0]      : 0.0f;
  const float bi1 = bias ? bias[n0 + 16] : 0.0f;
#pragma unroll
  for (int r = 0; r < 8; ++r) {
    const int m0 = tm + r + hlf * 8;
    C[(size_t)m0 * N + n0]             = (CT)(acc00[r] + bi0);
    C[(size_t)m0 * N + n0 + 16]        = (CT)(acc01[r] + bi1);
    C[(size_t)(m0 + 16) * N + n0]      = (CT)(acc10[r] + bi0);
    C[(size_t)(m0 + 16) * N + n0 + 16] = (CT)(acc11[r] + bi1);
  }
}

// ---- fused attention: one wave = (b, h, 16 query rows) -----------------
// S = (Q K^T) * 8 -> bf16 strip in LDS -> row softmax -> O = P V.
__global__ void __launch_bounds__(32)
attn_fused(const __bf16* __restrict__ qkv, __bf16* __restrict__ attn_out) {
  __shared__ __align__(16) __bf16 sS[16 * NSEQ];     // 32 KB score strip
  __shared__ __align__(16) __bf16 sV[2][32 * HDIM];  // 2 x 4 KB V stage

  const int lane = threadIdx.x & 31;
  const int m_l  = lane & 15;
  const int hlf  = lane >> 4;

  const int qt = blockIdx.x & 63;
  const int h  = (blockIdx.x >> 6) % NHEAD;
  const int b  = blockIdx.x / (64 * NHEAD);

  const size_t rstride = QKVD;  // 2304 bf16 per token
  const __bf16* base = qkv + (size_t)b * NSEQ * rstride + (size_t)h * HDIM;

  // persistent Q fragments (K-dim = HD = 64 = two k32 steps)
  const __bf16* qrow = base + (size_t)(qt * 16 + m_l) * rstride;
  const v16bf aQ0 = frag_a_bf(qrow, 0, lane);
  const v16bf aQ1 = frag_a_bf(qrow, 32, lane);

  // ---- phase 1: scores * 8 -> LDS ----
  for (int jt = 0; jt < NSEQ / 16; ++jt) {
    const __bf16* kcol = base + DMODEL + (size_t)(jt * 16 + m_l) * rstride;
    const v16bf bk0 = frag_b_bf(kcol, 0, lane);
    const v16bf bk1 = frag_b_bf(kcol, 32, lane);
    v8f s = {};
    s = WMMA_BF16(aQ0, bk0, s);
    s = WMMA_BF16(aQ1, bk1, s);
#pragma unroll
    for (int r = 0; r < 8; ++r)
      sS[(r + hlf * 8) * NSEQ + jt * 16 + m_l] = (__bf16)(s[r] * 8.0f);
  }
  __syncthreads();

  // ---- phase 2: softmax; lane = (row lane&15, half lane>>4), v8bf scans ----
  {
    const v8bf* rowv = (const v8bf*)&sS[m_l * NSEQ + hlf * (NSEQ / 2)];
    float mx = -1e30f;
    for (int c = 0; c < NSEQ / 16; ++c) {
      const v8bf v = rowv[c];
#pragma unroll
      for (int j = 0; j < 8; ++j) mx = fmaxf(mx, (float)v[j]);
    }
    mx = fmaxf(mx, __shfl_xor(mx, 16, 32));
    float sum = 0.0f;
    for (int c = 0; c < NSEQ / 16; ++c) {
      const v8bf v = rowv[c];
#pragma unroll
      for (int j = 0; j < 8; ++j) sum += __expf((float)v[j] - mx);
    }
    sum += __shfl_xor(sum, 16, 32);
    const float inv = 1.0f / sum;
    v8bf* roww = (v8bf*)&sS[m_l * NSEQ + hlf * (NSEQ / 2)];
    for (int c = 0; c < NSEQ / 16; ++c) {
      const v8bf v = rowv[c];
      v8bf p;
#pragma unroll
      for (int j = 0; j < 8; ++j) p[j] = (__bf16)(__expf((float)v[j] - mx) * inv);
      roww[c] = p;
    }
  }
  __syncthreads();

  // ---- phase 3: O = P V; V staged via async global->LDS, double-buffered ---
  // Lane copies V row (kt*32 + lane): 64 bf16 = 8 x async b128 (ASYNCcnt).
  const __bf16* vbase = base + 2 * DMODEL;

  auto issue_stage = [&](int kt, int buf) {
    const __bf16* vrow = vbase + (size_t)(kt * 32 + lane) * rstride;
    const unsigned lds0 = (unsigned)(size_t)&sV[buf][lane * HDIM];
#pragma unroll
    for (int c = 0; c < HDIM / 8; ++c) {
      asm volatile("global_load_async_to_lds_b128 %0, %1, off"
                   :: "v"(lds0 + c * 16), "v"(vrow + c * 8)
                   : "memory");
    }
  };

  issue_stage(0, 0);
  v8f o0 = {}, o1 = {}, o2 = {}, o3 = {};
  for (int kt = 0; kt < NSEQ / 32; ++kt) {
    const int buf = kt & 1;
    if (kt + 1 < NSEQ / 32) {
      issue_stage(kt + 1, buf ^ 1);
      // async loads complete in order: <=8 outstanding => current buf landed
      asm volatile("s_wait_asynccnt 0x8" ::: "memory");
    } else {
      asm volatile("s_wait_asynccnt 0x0" ::: "memory");
    }
    __syncthreads();

    // A frag from probability strip in LDS
    v16bf aP;
    {
      const int kbase = m_l * NSEQ + kt * 32 + hlf * 16;
      const v8bf lo = *(const v8bf*)&sS[kbase];
      const v8bf hi = *(const v8bf*)&sS[kbase + 8];
#pragma unroll
      for (int j = 0; j < 8; ++j) { aP[j] = lo[j]; aP[8 + j] = hi[j]; }
    }
    // B frags from staged V (scatter out of LDS, not global)
    const int kb = hlf * 16;
    v16bf bv0, bv1, bv2, bv3;
#pragma unroll
    for (int j = 0; j < 16; ++j) {
      const __bf16* vr = &sV[buf][(kb + j) * HDIM];
      bv0[j] = vr[0 * 16 + m_l];
      bv1[j] = vr[1 * 16 + m_l];
      bv2[j] = vr[2 * 16 + m_l];
      bv3[j] = vr[3 * 16 + m_l];
    }
    o0 = WMMA_BF16(aP, bv0, o0);
    o1 = WMMA_BF16(aP, bv1, o1);
    o2 = WMMA_BF16(aP, bv2, o2);
    o3 = WMMA_BF16(aP, bv3, o3);
  }

  // store O tile (bf16) to attn_out[b, qrow, h*64 + d]
#pragma unroll
  for (int r = 0; r < 8; ++r) {
    const int qr = qt * 16 + r + hlf * 8;
    __bf16* orow = attn_out + ((size_t)b * NSEQ + qr) * DMODEL + h * HDIM;
    orow[0 * 16 + m_l] = (__bf16)o0[r];
    orow[1 * 16 + m_l] = (__bf16)o1[r];
    orow[2 * 16 + m_l] = (__bf16)o2[r];
    orow[3 * 16 + m_l] = (__bf16)o3[r];
  }
}

extern "C" void kernel_launch(void* const* d_in, const int* in_sizes, int n_in,
                              void* d_out, int out_size, void* d_ws, size_t ws_size,
                              hipStream_t stream) {
  const float* x      = (const float*)d_in[0];   // [4,1024,768]
  const float* w_qkv  = (const float*)d_in[1];   // [2304,768]
  const float* w_proj = (const float*)d_in[2];   // [768,768]
  const float* b_proj = (const float*)d_in[3];   // [768]
  float* out = (float*)d_out;                    // [4,1024,768] f32

  const int M = BATCH * NSEQ;                    // 4096
  const int nx  = M * DMODEL;                    // 3,145,728
  const int nwq = QKVD * DMODEL;                 // 1,769,472
  const int nwp = DMODEL * DMODEL;               //   589,824

  __bf16* x_bf   = (__bf16*)d_ws;                //  6.29 MB
  __bf16* wq_bf  = x_bf  + nx;                   //  3.54 MB
  __bf16* wp_bf  = wq_bf + nwq;                  //  1.18 MB
  __bf16* qkv_bf = wp_bf + nwp;                  // 18.87 MB
  __bf16* att_bf = qkv_bf + (size_t)M * QKVD;    //  6.29 MB  (~36 MB total)

  // 0) one-shot conversions to the native WMMA operand format
  cvt_f32_to_bf16<<<nx  / (256 * 8), 256, 0, stream>>>(x,      x_bf,  nx);
  cvt_f32_to_bf16<<<nwq / (256 * 8), 256, 0, stream>>>(w_qkv,  wq_bf, nwq);
  cvt_f32_to_bf16<<<nwp / (256 * 8), 256, 0, stream>>>(w_proj, wp_bf, nwp);

  // 1) QKV projection: qkv = x @ w_qkv^T  (bf16 out)
  {
    const int waves = (M / 32) * (QKVD / 32);    // 9216
    wmma_gemm_nt<__bf16><<<waves / 4, 128, 0, stream>>>(
        x_bf, wq_bf, qkv_bf, nullptr, M, DMODEL, QKVD);
  }
  // 2) fused attention (scores*8 -> softmax -> @V), bf16 in/out
  {
    const int blocks = BATCH * NHEAD * (NSEQ / 16);  // 3072
    attn_fused<<<blocks, 32, 0, stream>>>(qkv_bf, att_bf);
  }
  // 3) output projection: out = attn @ w_proj^T + b_proj  (f32 out)
  {
    const int waves = (M / 32) * (DMODEL / 32);  // 3072
    wmma_gemm_nt<float><<<waves / 4, 128, 0, stream>>>(
        att_bf, wp_bf, out, b_proj, M, DMODEL, DMODEL);
  }
}
